// AttentionSE3_43009802502229
// MI455X (gfx1250) — matrix operand order repaired and checked by reading, not compiled, not run on hardware
//
#include <hip/hip_runtime.h>

typedef __attribute__((ext_vector_type(16))) _Float16 v16h;
typedef __attribute__((ext_vector_type(8)))  float    v8f;

#define NPTS 6144   // N*K = 256*24
#define MID  128
#define NN   256
#define KK   24
#define HID  64

// ---------------------------------------------------------------------------
// Kernel 0: one-shot weight prep.  src: f32 [128][ncols] (row-major, K-major)
//           dst: f16 [ncols][128]  (so B-fragment K-runs are contiguous)
// ---------------------------------------------------------------------------
__global__ __launch_bounds__(256) void cvt_transpose_kernel(
    const float* __restrict__ src, _Float16* __restrict__ dst, int ncols)
{
  const int idx = blockIdx.x * 256 + threadIdx.x;
  if (idx >= ncols * MID) return;
  const int col = idx >> 7;          // /128
  const int k   = idx & 127;
  dst[idx] = (_Float16)src[(size_t)k * ncols + col];
}

// ---------------------------------------------------------------------------
// Kernel 1: per-network radial MLP front-end.
//   h1 = relu(LN(rd*W1 + b1));  h2 = relu(LN(h1 @ W2 + b2))   -> f16 in ws
// Block: 256 threads = 8 waves, one network x 16 points.
// W2 GEMM: 8 column tiles (one per wave), 4x wmma_f32_16x16x32_f16, K=128.
// ---------------------------------------------------------------------------
__global__ __launch_bounds__(256) void radial_h_kernel(
    const float* __restrict__ rel_dist,
    const float* __restrict__ rn_W1, const float* __restrict__ rn_b1,
    const float* __restrict__ rn_g1, const float* __restrict__ rn_be1,
    const _Float16* __restrict__ W2T,            // [net][col][k] f16
    const float* __restrict__ rn_b2,
    const float* __restrict__ rn_g2, const float* __restrict__ rn_be2,
    _Float16* __restrict__ h2out)
{
  const int net  = blockIdx.x;           // 0..7
  const int tile = blockIdx.y;           // 0..383
  const int tid  = threadIdx.x;
  const int lane = tid & 31;
  const int wave = tid >> 5;
  const int pt   = tid >> 4;             // point in tile (16 threads/point)
  const int sub  = tid & 15;
  const int g0   = tile * 16;

  __shared__ _Float16 h1[16][MID];
  __shared__ float    h2raw[16][MID];
  __shared__ float    redA[16][16], redB[16][16];
  __shared__ float    stat[16][2];

  // ---- stage A: h1 = relu(LN(rd*W1 + b1)) ----
  const float rd  = rel_dist[g0 + pt];
  const float* W1 = rn_W1 + net * MID;
  const float* b1 = rn_b1 + net * MID;
  float vals[8];
  float s = 0.f, s2 = 0.f;
#pragma unroll
  for (int j = 0; j < 8; ++j) {
    const int m = sub * 8 + j;
    const float v = rd * W1[m] + b1[m];
    vals[j] = v; s += v; s2 += v * v;
  }
  redA[pt][sub] = s; redB[pt][sub] = s2;
  __syncthreads();
  if (sub == 0) {
    float ts = 0.f, ts2 = 0.f;
    for (int u = 0; u < 16; ++u) { ts += redA[pt][u]; ts2 += redB[pt][u]; }
    const float mu = ts * (1.0f / MID);
    stat[pt][0] = mu;
    stat[pt][1] = rsqrtf(ts2 * (1.0f / MID) - mu * mu + 1e-5f);
  }
  __syncthreads();
  {
    const float mu = stat[pt][0], rs = stat[pt][1];
    const float* g1  = rn_g1  + net * MID;
    const float* be1 = rn_be1 + net * MID;
#pragma unroll
    for (int j = 0; j < 8; ++j) {
      const int m = sub * 8 + j;
      const float v = (vals[j] - mu) * rs * g1[m] + be1[m];
      h1[pt][m] = (_Float16)fmaxf(v, 0.f);
    }
  }
  __syncthreads();

  // ---- stage B: h2raw = h1 @ W2 + b2 via WMMA ----
  v8f c = {};
  const int rowA = lane & 15;
  const int kAhi = (lane >= 16) ? 16 : 0;
  const int colB = wave * 16 + (lane & 15);
  const int kBhi = (lane >= 16) ? 16 : 0;
  const _Float16* W2c = W2T + ((size_t)net * MID + colB) * MID;  // this col's K-run
#pragma unroll
  for (int kk = 0; kk < 4; ++kk) {
    v16h a, bf;
#pragma unroll
    for (int j = 0; j < 16; ++j)
      a[j] = h1[rowA][kk * 32 + kAhi + j];
#pragma unroll
    for (int j = 0; j < 16; ++j)
      bf[j] = W2c[kk * 32 + kBhi + j];              // contiguous f16 -> b128
    c = __builtin_amdgcn_wmma_f32_16x16x32_f16(false, a, false, bf,
                                               (short)0, c, false, false);
  }
  {
    const float* b2 = rn_b2 + net * MID;
    const int mb = (lane >= 16) ? 8 : 0;
#pragma unroll
    for (int r = 0; r < 8; ++r)
      h2raw[mb + r][colB] = c[r] + b2[colB];
  }
  __syncthreads();

  // ---- stage C: LN + relu + store f16 ----
  {
    float ts = 0.f, ts2 = 0.f;
#pragma unroll
    for (int j = 0; j < 8; ++j) {
      const float v = h2raw[pt][sub * 8 + j];
      ts += v; ts2 += v * v;
    }
    redA[pt][sub] = ts; redB[pt][sub] = ts2;
  }
  __syncthreads();
  if (sub == 0) {
    float ts = 0.f, ts2 = 0.f;
    for (int u = 0; u < 16; ++u) { ts += redA[pt][u]; ts2 += redB[pt][u]; }
    const float mu = ts * (1.0f / MID);
    stat[pt][0] = mu;
    stat[pt][1] = rsqrtf(ts2 * (1.0f / MID) - mu * mu + 1e-5f);
  }
  __syncthreads();
  {
    const float mu = stat[pt][0], rs = stat[pt][1];
    const float* g2  = rn_g2  + net * MID;
    const float* be2 = rn_be2 + net * MID;
    _Float16* dst = h2out + ((size_t)net * NPTS + g0 + pt) * MID;
#pragma unroll
    for (int j = 0; j < 8; ++j) {
      const int m = sub * 8 + j;
      const float v = (h2raw[pt][m] - mu) * rs * g2[m] + be2[m];
      dst[m] = (_Float16)fmaxf(v, 0.f);
    }
  }
}

// ---------------------------------------------------------------------------
// Kernel 2: fused  R = h2 @ W3 + b3  (WMMA)  with per-point contraction
//   acc[pt,o,p] += sum_{i,f} R[pt,(o,i,f)] * T[pt,i,f,p],
//   T[pt,i,f,p]  = sum_q basis[pt,p,q,f] * feat_di[nbr[pt], i, q]
// Basis slab for the tile is staged into LDS with the gfx1250 async engine
// (global_load_async_to_lds_b128 + s_wait_asynccnt).  Each 16-column WMMA slab
// lies inside a single o, so the lane dim is a pure 16-wide xor-butterfly
// reduction; lanes 0/16 issue one conflict-free LDS atomic per (row,p).
// Grid: (384 point-tiles, 2 do).  One launch per branch (k / v).
// ---------------------------------------------------------------------------
__global__ __launch_bounds__(256) void conv_kernel(
    const _Float16* __restrict__ h2,
    const float* __restrict__ feat0,   const float* __restrict__ feat1,
    const float* __restrict__ basis00, const float* __restrict__ basis01,
    const float* __restrict__ basis10, const float* __restrict__ basis11,
    const int*   __restrict__ nbr_idx,
    const _Float16* __restrict__ W3T_00, const float* __restrict__ b3_00,
    const _Float16* __restrict__ W3T_01, const float* __restrict__ b3_01,
    const _Float16* __restrict__ W3T_10, const float* __restrict__ b3_10,
    const _Float16* __restrict__ W3T_11, const float* __restrict__ b3_11,
    int net_off,
    float* __restrict__ out_do0, float* __restrict__ out_do1)
{
  const int tile = blockIdx.x;          // 0..383
  const int dO   = blockIdx.y;          // 0/1
  const int tid  = threadIdx.x;
  const int lane = tid & 31;
  const int wave = tid >> 5;
  const int p    = 2 * dO + 1;

  __shared__ float acc[16][HID][3];     // [pt][o][p]
  __shared__ float T[16][16][3][3];     // [pt][i][f][p]
  __shared__ __align__(16) float basL[16 * 27];   // basis slab (max pqnf=27)
  __shared__ int   nb[16];

  for (int idx = tid; idx < 16 * HID * 3; idx += 256)
    ((float*)acc)[idx] = 0.f;
  if (tid < 16) nb[tid] = nbr_idx[tile * 16 + tid];
  __syncthreads();

  for (int di = 0; di < 2; ++di) {
    const int q  = 2 * di + 1;
    const int nf = 2 * ((di < dO) ? di : dO) + 1;
    const float* feat = di ? feat1 : feat0;
    const float* bas  = (di == 0) ? (dO == 0 ? basis00 : basis01)
                                  : (dO == 0 ? basis10 : basis11);
    const _Float16* W3T = (di == 0) ? (dO == 0 ? W3T_00 : W3T_01)
                                    : (dO == 0 ? W3T_10 : W3T_11);
    const float* b3   = (di == 0) ? (dO == 0 ? b3_00 : b3_01)
                                  : (dO == 0 ? b3_10 : b3_11);
    const _Float16* A = h2 +
        ((size_t)(net_off + 2 * di + dO) * NPTS + (size_t)tile * 16) * MID;
    const int pqnf = p * q * nf;

    // ---- async-stage the contiguous basis slab for this (tile,di) into LDS.
    // Generic LDS-aperture addresses carry the LDS byte offset in their low
    // 32 bits, so (unsigned)(size_t)sharedPtr is the async VDST value.
    {
      const unsigned nchunk  = (unsigned)(16 * pqnf * 4) >> 4;   // 16B chunks
      const unsigned ldsbase = (unsigned)(size_t)(const void*)basL;
      const unsigned long long gbase =
          (unsigned long long)(size_t)(const void*)(bas + (size_t)(tile * 16) * pqnf);
      if ((unsigned)tid < nchunk) {
        const unsigned loff = ldsbase + (unsigned)tid * 16u;
        const unsigned long long ga = gbase + (unsigned long long)tid * 16u;
        asm volatile("global_load_async_to_lds_b128 %0, %1, off"
                     :: "v"(loff), "v"(ga) : "memory");
      }
      asm volatile("s_wait_asynccnt 0x0" ::: "memory");
    }
    __syncthreads();

    // ---- build T table for this di (basis read from LDS) ----
    const int totT = 16 * 16 * nf * p;
    for (int idx = tid; idx < totT; idx += 256) {
      const int pt = idx / (16 * nf * p);
      int rem = idx % (16 * nf * p);
      const int i  = rem / (nf * p);  rem %= (nf * p);
      const int f  = rem / p;
      const int pi = rem % p;
      const float* x  = feat + (size_t)nb[pt] * (16 * q) + i * q;
      const float* bb = basL + pt * pqnf + pi * (q * nf) + f;
      float sT = 0.f;
      for (int qq = 0; qq < q; ++qq) sT += bb[qq * nf] * x[qq];
      T[pt][i][f][pi] = sT;
    }
    __syncthreads();

    // ---- load A fragments once, reuse across all column tiles ----
    v16h afr[4];
    const int rowA = lane & 15;
    const int kAhi = (lane >= 16) ? 16 : 0;
#pragma unroll
    for (int kk = 0; kk < 4; ++kk)
#pragma unroll
      for (int j = 0; j < 16; ++j)
        afr[kk][j] = A[(size_t)rowA * MID + kk * 32 + kAhi + j];

    const int od     = 1024 * nf;       // HID*DIM*nf
    const int ntiles = od >> 4;
    const int colB   = lane & 15;
    const int kBhi   = (lane >= 16) ? 16 : 0;
    const int mb     = (lane >= 16) ? 8 : 0;

    for (int ct = wave; ct < ntiles; ct += 8) {
      const int cbase = ct * 16;
      // speculative prefetch: this wave's next weight slab (ct+8 -> +128 cols)
      if (ct + 8 < ntiles)
        __builtin_prefetch((const void*)(W3T + (size_t)(cbase + 128) * MID), 0, 1);
      v8f c = {};
      const _Float16* Wc = W3T + (size_t)(cbase + colB) * MID;   // this col's K-run
#pragma unroll
      for (int kk = 0; kk < 4; ++kk) {
        v16h bf;
#pragma unroll
        for (int j = 0; j < 16; ++j)
          bf[j] = Wc[kk * 32 + kBhi + j];          // contiguous f16 -> b128
        c = __builtin_amdgcn_wmma_f32_16x16x32_f16(false, afr[kk], false, bf,
                                                   (short)0, c, false, false);
      }
      // contract this 16-wide column slab against T
      const int cgl = cbase + colB;               // global column = (o,i,f)
      const int o_tile = cbase / (16 * nf);       // uniform across the slab
      const int rem = cgl % (16 * nf);
      const int ii = rem / nf;
      const int ff = rem % nf;
      const float bias = b3[cgl];

      float part[8][3];
#pragma unroll
      for (int r = 0; r < 8; ++r) {
        const float val = c[r] + bias;
#pragma unroll
        for (int pi = 0; pi < 3; ++pi)
          if (pi < p) part[r][pi] = val * T[mb + r][ii][ff][pi];
      }
      // xor-butterfly reduction across the 16 columns (stays in each half)
#pragma unroll
      for (int st = 1; st <= 8; st <<= 1)
#pragma unroll
        for (int r = 0; r < 8; ++r)
#pragma unroll
          for (int pi = 0; pi < 3; ++pi)
            if (pi < p) part[r][pi] += __shfl_xor(part[r][pi], st, 32);
      if ((lane & 15) == 0) {
#pragma unroll
        for (int r = 0; r < 8; ++r)
#pragma unroll
          for (int pi = 0; pi < 3; ++pi)
            if (pi < p) atomicAdd(&acc[mb + r][o_tile][pi], part[r][pi]);
      }
    }
    __syncthreads();
  }

  // ---- write accumulated conv output ----
  float* dst = dO ? out_do1 : out_do0;
  const int tot = 16 * HID * p;
  for (int idx = tid; idx < tot; idx += 256) {
    const int pt = idx / (HID * p);
    int rem = idx % (HID * p);
    const int o  = rem / p;
    const int pi = rem % p;
    dst[((size_t)(tile * 16 + pt)) * HID * p + o * p + pi] = acc[pt][o][pi];
  }
}

// ---------------------------------------------------------------------------
// Kernel 3: attention over K=24 neighbors + output projection.
// Grid: (N, 2 degrees). Tiny problem -> plain VALU kernel.
// ---------------------------------------------------------------------------
__global__ __launch_bounds__(256) void attn_kernel(
    const float* __restrict__ feat0, const float* __restrict__ feat1,
    const float* __restrict__ Wq0,   const float* __restrict__ Wq1,
    const float* __restrict__ Wout0, const float* __restrict__ Wout1,
    const float* __restrict__ k0, const float* __restrict__ k1,
    const float* __restrict__ v0, const float* __restrict__ v1,
    const unsigned char* __restrict__ mask,
    float* __restrict__ out)
{
  const int n   = blockIdx.x;
  const int deg = blockIdx.y;
  const int tid = threadIdx.x;
  const int m   = 2 * deg + 1;
  const float* feat = deg ? feat1 : feat0;
  const float* Wq   = deg ? Wq1   : Wq0;
  const float* Wout = deg ? Wout1 : Wout0;
  const float* kd   = deg ? k1 : k0;
  const float* vd   = deg ? v1 : v0;

  __shared__ float qL[HID][3];
  __shared__ float simL[4][KK];
  __shared__ float attnL[4][KK];
  __shared__ float ohL[HID][3];

  if (tid < HID * m) {
    const int e = tid / m, mm = tid % m;
    float sq = 0.f;
    for (int d = 0; d < 16; ++d)
      sq += feat[(size_t)n * 16 * m + d * m + mm] * Wq[d * HID + e];
    qL[e][mm] = sq;
  }
  __syncthreads();

  if (tid < 4 * KK) {
    const int h = tid / KK, j = tid % KK;
    const float* kp = kd + ((size_t)(n * KK + j)) * HID * m + (h * 16) * m;
    float sv = 0.f;
    for (int d = 0; d < 16; ++d)
      for (int mm = 0; mm < m; ++mm)
        sv += qL[h * 16 + d][mm] * kp[d * m + mm];
    sv *= 0.25f;                               // DIM_HEAD^-0.5
    if (!mask[n * KK + j]) sv = -3.402823466e38f;
    simL[h][j] = sv;
  }
  __syncthreads();

  if (tid < 4) {
    const int h = tid;
    float mx = -3.402823466e38f;
    for (int j = 0; j < KK; ++j) mx = fmaxf(mx, simL[h][j]);
    float sm = 0.f;
    for (int j = 0; j < KK; ++j) {
      const float e = __expf(simL[h][j] - mx);
      attnL[h][j] = e; sm += e;
    }
    const float inv = 1.f / sm;
    for (int j = 0; j < KK; ++j) attnL[h][j] *= inv;
  }
  __syncthreads();

  if (tid < HID * m) {
    const int e = tid / m, mm = tid % m, h = e >> 4;
    float so = 0.f;
    for (int j = 0; j < KK; ++j)
      so += attnL[h][j] * vd[((size_t)(n * KK + j)) * HID * m + e * m + mm];
    ohL[e][mm] = so;
  }
  __syncthreads();

  if (tid < 16 * m) {
    const int d = tid / m, mm = tid % m;
    float sr = 0.f;
    for (int e = 0; e < HID; ++e) sr += ohL[e][mm] * Wout[e * 16 + d];
    float* dst = out + (deg ? 4096 : 0);      // out0 then out1, flat
    dst[(size_t)n * 16 * m + d * m + mm] = sr;
  }
}

// ---------------------------------------------------------------------------
extern "C" void kernel_launch(void* const* d_in, const int* in_sizes, int n_in,
                              void* d_out, int out_size, void* d_ws, size_t ws_size,
                              hipStream_t stream) {
  (void)in_sizes; (void)n_in; (void)out_size; (void)ws_size;
  const float* feat0   = (const float*)d_in[0];
  const float* feat1   = (const float*)d_in[1];
  const float* reldist = (const float*)d_in[2];
  const float* basis00 = (const float*)d_in[3];
  const float* basis01 = (const float*)d_in[4];
  const float* basis10 = (const float*)d_in[5];
  const float* basis11 = (const float*)d_in[6];
  const float* Wq0     = (const float*)d_in[7];
  const float* Wq1     = (const float*)d_in[8];
  const float* Wout0   = (const float*)d_in[9];
  const float* Wout1   = (const float*)d_in[10];
  const float* rn_W1   = (const float*)d_in[11];
  const float* rn_b1   = (const float*)d_in[12];
  const float* rn_g1   = (const float*)d_in[13];
  const float* rn_be1  = (const float*)d_in[14];
  const float* rn_W2   = (const float*)d_in[15];
  const float* rn_b2   = (const float*)d_in[16];
  const float* rn_g2   = (const float*)d_in[17];
  const float* rn_be2  = (const float*)d_in[18];
  // dict order: W3/b3 interleaved, k-branch then v-branch, pairs 00,01,10,11
  const float* W3k[4] = {(const float*)d_in[19], (const float*)d_in[21],
                         (const float*)d_in[23], (const float*)d_in[25]};
  const float* b3k[4] = {(const float*)d_in[20], (const float*)d_in[22],
                         (const float*)d_in[24], (const float*)d_in[26]};
  const float* W3v[4] = {(const float*)d_in[27], (const float*)d_in[29],
                         (const float*)d_in[31], (const float*)d_in[33]};
  const float* b3v[4] = {(const float*)d_in[28], (const float*)d_in[30],
                         (const float*)d_in[32], (const float*)d_in[34]};
  const int* nbr_idx = (const int*)d_in[35];
  const unsigned char* nbr_mask = (const unsigned char*)d_in[36];

  // workspace layout (all chunks 256B-aligned by construction)
  char* ws = (char*)d_ws;
  _Float16* h2  = (_Float16*)ws;                 ws += (size_t)8 * NPTS * MID * 2;
  _Float16* W2T = (_Float16*)ws;                 ws += (size_t)8 * MID * MID * 2;
  static const int ODS[4] = {1024, 1024, 1024, 3072};   // od per pair 00,01,10,11
  _Float16* W3Tk[4]; _Float16* W3Tv[4];
  for (int pr = 0; pr < 4; ++pr) { W3Tk[pr] = (_Float16*)ws; ws += (size_t)ODS[pr] * MID * 2; }
  for (int pr = 0; pr < 4; ++pr) { W3Tv[pr] = (_Float16*)ws; ws += (size_t)ODS[pr] * MID * 2; }
  float* conv = (float*)ws;
  const size_t S0 = (size_t)NPTS * HID;          // do=0 (p=1)
  const size_t S1 = (size_t)NPTS * HID * 3;      // do=1 (p=3)
  float* k0 = conv;
  float* k1 = k0 + S0;
  float* v0 = k1 + S1;
  float* v1 = v0 + S0;

  // ---- weight prep: f32 row-major -> f16 [col][k] ----
  for (int net = 0; net < 8; ++net)
    cvt_transpose_kernel<<<(MID * MID) / 256, 256, 0, stream>>>(
        rn_W2 + (size_t)net * MID * MID, W2T + (size_t)net * MID * MID, MID);
  for (int pr = 0; pr < 4; ++pr) {
    cvt_transpose_kernel<<<(ODS[pr] * MID) / 256, 256, 0, stream>>>(
        W3k[pr], W3Tk[pr], ODS[pr]);
    cvt_transpose_kernel<<<(ODS[pr] * MID) / 256, 256, 0, stream>>>(
        W3v[pr], W3Tv[pr], ODS[pr]);
  }

  dim3 g1(8, NPTS / 16);
  radial_h_kernel<<<g1, 256, 0, stream>>>(
      reldist, rn_W1, rn_b1, rn_g1, rn_be1, W2T, rn_b2, rn_g2, rn_be2, h2);

  dim3 g2(NPTS / 16, 2);
  conv_kernel<<<g2, 256, 0, stream>>>(
      h2, feat0, feat1, basis00, basis01, basis10, basis11, nbr_idx,
      W3Tk[0], b3k[0], W3Tk[1], b3k[1], W3Tk[2], b3k[2], W3Tk[3], b3k[3],
      /*net_off=*/0, k0, k1);
  conv_kernel<<<g2, 256, 0, stream>>>(
      h2, feat0, feat1, basis00, basis01, basis10, basis11, nbr_idx,
      W3Tv[0], b3v[0], W3Tv[1], b3v[1], W3Tv[2], b3v[2], W3Tv[3], b3v[3],
      /*net_off=*/4, v0, v1);

  dim3 g3(NN, 2);
  attn_kernel<<<g3, 256, 0, stream>>>(
      feat0, feat1, Wq0, Wq1, Wout0, Wout1, k0, k1, v0, v1, nbr_mask,
      (float*)d_out);
}
